// GraphAutoencoder_48808008352178
// MI455X (gfx1250) — compile-verified, matrix operand order checked by reading, not run on hardware
//
#include <hip/hip_runtime.h>

// ---------------------------------------------------------------------------
// Types for CDNA5 WMMA (wave32)
// ---------------------------------------------------------------------------
typedef __bf16 bf16_t;
typedef bf16_t v16bf __attribute__((ext_vector_type(16)));
typedef bf16_t v8bf  __attribute__((ext_vector_type(8)));
typedef float  v8f   __attribute__((ext_vector_type(8)));

__device__ __forceinline__ unsigned short f2bf(float f) {
    unsigned u = __float_as_uint(f);
    unsigned r = u + 0x7FFFu + ((u >> 16) & 1u);   // round to nearest even
    return (unsigned short)(r >> 16);
}

// ---------------------------------------------------------------------------
// fp32 -> bf16 convert
// ---------------------------------------------------------------------------
__global__ __launch_bounds__(256)
void cvt_bf16_k(const float* __restrict__ in, unsigned short* __restrict__ out, long n) {
    long t = (long)blockIdx.x * 256 + threadIdx.x;
    if (t < n) out[t] = f2bf(in[t]);
}

// W [nMat][K][Ncol] fp32 row-major  ->  Wt [nMat][Ncol][K] bf16
__global__ __launch_bounds__(256)
void transpose_bf16_k(const float* __restrict__ W, unsigned short* __restrict__ Wt,
                      int K, int Ncol, int total) {
    int t = blockIdx.x * 256 + threadIdx.x;
    if (t >= total) return;
    int kn  = K * Ncol;
    int m   = t / kn;
    int rem = t - m * kn;
    int k   = rem / Ncol;
    int n   = rem - k * Ncol;
    Wt[(size_t)m * kn + (size_t)n * K + k] = f2bf(W[t]);
}

// ---------------------------------------------------------------------------
// GEMM: C[rows x 128] = A(bf16, rows x K) @ Bt(bf16, 128 x K)^T  (+bias, relu)
// Block = 256 threads = 8 waves; each wave computes a 16-row x 128-col strip.
// B panel staged in LDS (shared by all 8 waves).  grid.y = head index.
// ---------------------------------------------------------------------------
__global__ __launch_bounds__(256)
void gemm_wmma_bf16_k(const unsigned short* __restrict__ A,
                      const unsigned short* __restrict__ Bt,
                      float* __restrict__ C,
                      const float* __restrict__ bias,
                      int rows, int K, int ldc,
                      long aHeadStride, long btHeadStride, long cHeadStride,
                      int doRelu) {
    extern __shared__ unsigned char smem_raw[];
    bf16_t* bs = (bf16_t*)smem_raw;

    const int h = blockIdx.y;
    const bf16_t* Ah  = (const bf16_t*)A  + (size_t)h * aHeadStride;
    const bf16_t* Bth = (const bf16_t*)Bt + (size_t)h * btHeadStride;
    float* Ch = C + (size_t)h * cHeadStride;

    // Cooperative copy of the 128 x K B panel into LDS (16B vectors).
    {
        const v8bf* g = (const v8bf*)Bth;
        v8bf* s = (v8bf*)bs;
        int nvec = (K * 128) / 8;
        for (int i = threadIdx.x; i < nvec; i += 256) s[i] = g[i];
    }
    __syncthreads();

    const int wave = threadIdx.x >> 5;
    const int lane = threadIdx.x & 31;
    const int row0 = (blockIdx.x * 8 + wave) * 16;
    if (row0 < rows) {
        const int mrow = lane & 15;        // M (for A) / N (for B) sub-index
        const int ko   = (lane >> 4) * 8;  // K sub-offset per ISA 16-bit layout

        const bf16_t* arow = Ah + (size_t)(row0 + mrow) * K;

        v8f acc[8] = {};
        for (int k0 = 0; k0 < K; k0 += 32) {
            v8bf alo = *(const v8bf*)(arow + k0 + ko);
            v8bf ahi = *(const v8bf*)(arow + k0 + 16 + ko);
            v16bf a = __builtin_shufflevector(alo, ahi,
                        0,1,2,3,4,5,6,7,8,9,10,11,12,13,14,15);
            #pragma unroll
            for (int nt = 0; nt < 8; ++nt) {
                const bf16_t* brow = bs + (size_t)(nt * 16 + mrow) * K + k0 + ko;
                v8bf blo = *(const v8bf*)(brow);
                v8bf bhi = *(const v8bf*)(brow + 16);
                v16bf b = __builtin_shufflevector(blo, bhi,
                            0,1,2,3,4,5,6,7,8,9,10,11,12,13,14,15);
                acc[nt] = __builtin_amdgcn_wmma_f32_16x16x32_bf16(
                            false, a, false, b, (short)0, acc[nt], false, false);
            }
        }

        // Epilogue: D layout — lane 0-15: row = row0 + v, col = lane;
        //                      lane 16-31: row = row0 + 8 + v, col = lane-16.
        const int rbase = row0 + 8 * (lane >> 4);
        #pragma unroll
        for (int nt = 0; nt < 8; ++nt) {
            int col = nt * 16 + mrow;
            float bv = bias ? bias[col] : 0.0f;
            #pragma unroll
            for (int v = 0; v < 8; ++v) {
                float val = acc[nt][v] + bv;
                if (doRelu) val = fmaxf(val, 0.0f);
                Ch[(size_t)(rbase + v) * ldc + col] = val;
            }
        }
    }
}

// ---------------------------------------------------------------------------
// Per-row attention scores: s_src = H . a_src, s_dst = H . a_dst
// One wave per (head, row); float4 loads + xor-shuffle reduction (wave32).
// ---------------------------------------------------------------------------
__global__ __launch_bounds__(256)
void row_dot_k(const float* __restrict__ H, const float* __restrict__ asrc,
               const float* __restrict__ adst, float* __restrict__ ssrc,
               float* __restrict__ sdst, int N) {
    int h   = blockIdx.y;
    int row = blockIdx.x * 8 + (threadIdx.x >> 5);
    if (row >= N) return;
    int lane = threadIdx.x & 31;
    const float4* hp = (const float4*)(H + ((size_t)h * N + row) * 128);
    const float4* as = (const float4*)(asrc + h * 128);
    const float4* ad = (const float4*)(adst + h * 128);
    float4 x = hp[lane], a = as[lane], b = ad[lane];
    float p1 = x.x * a.x + x.y * a.y + x.z * a.z + x.w * a.w;
    float p2 = x.x * b.x + x.y * b.y + x.z * b.z + x.w * b.w;
    #pragma unroll
    for (int m = 16; m >= 1; m >>= 1) {
        p1 += __shfl_xor(p1, m, 32);
        p2 += __shfl_xor(p2, m, 32);
    }
    if (lane == 0) { ssrc[h * N + row] = p1; sdst[h * N + row] = p2; }
}

__global__ __launch_bounds__(256)
void zero2_k(float* __restrict__ denom, unsigned* __restrict__ mkey, int total) {
    int t = blockIdx.x * 256 + threadIdx.x;
    if (t < total) { denom[t] = 0.0f; mkey[t] = 0u; }
}

__global__ __launch_bounds__(256)
void init_cat_k(float* __restrict__ cat, const float* __restrict__ bflat, long total) {
    long t = (long)blockIdx.x * 256 + threadIdx.x;
    if (t < total) cat[t] = bflat[(int)(t & 511)];
}

// monotone float<->u32 encoding for atomicMax-based segment max
__device__ __forceinline__ unsigned enc_f(float v) {
    unsigned u = __float_as_uint(v);
    return (u & 0x80000000u) ? ~u : (u | 0x80000000u);
}
__device__ __forceinline__ float dec_f(unsigned u) {
    return __uint_as_float((u & 0x80000000u) ? (u & 0x7FFFFFFFu) : ~u);
}

__global__ __launch_bounds__(256)
void edge_max_k(const int* __restrict__ src, const int* __restrict__ dst,
                const float* __restrict__ ssrc, const float* __restrict__ sdst,
                unsigned* __restrict__ mkey, int E, int N, int total) {
    int t = blockIdx.x * 256 + threadIdx.x;
    if (t >= total) return;
    int h = t & 3, e = t >> 2;
    int s, d;
    if (e < E) { s = src[e]; d = dst[e]; } else { s = d = e - E; }
    float v = ssrc[h * N + s] + sdst[h * N + d];
    v = v > 0.0f ? v : 0.2f * v;                 // leaky_relu(0.2)
    atomicMax(&mkey[h * N + d], enc_f(v));
}

__global__ __launch_bounds__(256)
void edge_exp_k(const int* __restrict__ src, const int* __restrict__ dst,
                const float* __restrict__ ssrc, const float* __restrict__ sdst,
                const unsigned* __restrict__ mkey, float* __restrict__ denom,
                float* __restrict__ ew, int E, int N, int total) {
    int t = blockIdx.x * 256 + threadIdx.x;
    if (t >= total) return;
    int h = t & 3, e = t >> 2;
    int s, d;
    if (e < E) { s = src[e]; d = dst[e]; } else { s = d = e - E; }
    float v = ssrc[h * N + s] + sdst[h * N + d];
    v = v > 0.0f ? v : 0.2f * v;
    float w = __expf(v - dec_f(mkey[h * N + d]));
    ew[t] = w;
    atomicAdd(&denom[h * N + d], w);
}

// One wave per (edge, head): cat[dst, h*128 + c] += alpha * H[h][src][c]
__global__ __launch_bounds__(256)
void edge_scatter_k(const int* __restrict__ src, const int* __restrict__ dst,
                    const float* __restrict__ ew, const float* __restrict__ denom,
                    const float* __restrict__ H, float* __restrict__ cat,
                    int E, int N, int pairs) {
    int p = blockIdx.x * 8 + (threadIdx.x >> 5);
    if (p >= pairs) return;
    int lane = threadIdx.x & 31;
    int h = p & 3, e = p >> 2;
    int s, d;
    if (e < E) { s = src[e]; d = dst[e]; } else { s = d = e - E; }
    float alpha = ew[p] / denom[h * N + d];
    const float4* hp = (const float4*)(H + ((size_t)h * N + s) * 128);
    float4 x = hp[lane];
    float* ob = cat + (size_t)d * 512 + h * 128 + lane * 4;
    atomicAdd(ob + 0, alpha * x.x);
    atomicAdd(ob + 1, alpha * x.y);
    atomicAdd(ob + 2, alpha * x.z);
    atomicAdd(ob + 3, alpha * x.w);
}

// ---------------------------------------------------------------------------
// Host orchestration
// ---------------------------------------------------------------------------
extern "C" void kernel_launch(void* const* d_in, const int* in_sizes, int n_in,
                              void* d_out, int out_size, void* d_ws, size_t ws_size,
                              hipStream_t stream) {
    (void)in_sizes; (void)n_in; (void)out_size; (void)ws_size;

    constexpr int N = 50000, E = 800000, Etot = E + N, D = 128;

    const float* x         = (const float*)d_in[0];
    const int*   ei        = (const int*)  d_in[1];
    const float* enc_W     = (const float*)d_in[2];
    const float* enc_asrc  = (const float*)d_in[3];
    const float* enc_adst  = (const float*)d_in[4];
    const float* enc_b     = (const float*)d_in[5];
    const float* enc_lin_W = (const float*)d_in[6];
    const float* enc_lin_b = (const float*)d_in[7];
    const float* dec_W     = (const float*)d_in[8];
    const float* dec_asrc  = (const float*)d_in[9];
    const float* dec_adst  = (const float*)d_in[10];
    const float* dec_b     = (const float*)d_in[11];
    const float* dec_lin_W = (const float*)d_in[12];
    const float* dec_lin_b = (const float*)d_in[13];
    float* out = (float*)d_out;

    const int* srcA = ei;
    const int* dstA = ei + E;

    // ---- carve workspace ----
    char* ws = (char*)d_ws; size_t off = 0;
    auto carve = [&](size_t bytes) -> void* {
        void* p = ws + off;
        off = (off + bytes + 255) & ~(size_t)255;
        return p;
    };
    unsigned short* Xbf  = (unsigned short*)carve((size_t)N * D * 2);
    unsigned short* WtE  = (unsigned short*)carve((size_t)4 * 128 * 128 * 2);
    unsigned short* LWtE = (unsigned short*)carve((size_t)128 * 512 * 2);
    unsigned short* WtD  = (unsigned short*)carve((size_t)4 * 128 * 128 * 2);
    unsigned short* LWtD = (unsigned short*)carve((size_t)128 * 512 * 2);
    float*    Hbuf  = (float*)   carve((size_t)4 * N * D * 4);
    float*    ssrc  = (float*)   carve((size_t)4 * N * 4);
    float*    sdst  = (float*)   carve((size_t)4 * N * 4);
    float*    denom = (float*)   carve((size_t)4 * N * 4);
    unsigned* mkey  = (unsigned*)carve((size_t)4 * N * 4);
    float*    ew    = (float*)   carve((size_t)4 * Etot * 4);
    float*    cat   = (float*)   carve((size_t)N * 512 * 4);
    unsigned short* catbf = (unsigned short*)carve((size_t)N * 512 * 2);
    float*    Yenc  = (float*)   carve((size_t)N * D * 4);
    unsigned short* Ybf = (unsigned short*)carve((size_t)N * D * 2);

    // ---- preprocess: bf16 conversions / weight transposes ----
    cvt_bf16_k<<<(N * D + 255) / 256, 256, 0, stream>>>(x, Xbf, (long)N * D);
    transpose_bf16_k<<<(4 * 128 * 128 + 255) / 256, 256, 0, stream>>>(enc_W, WtE, 128, 128, 4 * 128 * 128);
    transpose_bf16_k<<<(512 * 128 + 255) / 256, 256, 0, stream>>>(enc_lin_W, LWtE, 512, 128, 512 * 128);
    transpose_bf16_k<<<(4 * 128 * 128 + 255) / 256, 256, 0, stream>>>(dec_W, WtD, 128, 128, 4 * 128 * 128);
    transpose_bf16_k<<<(512 * 128 + 255) / 256, 256, 0, stream>>>(dec_lin_W, LWtD, 512, 128, 512 * 128);

    auto run_layer = [&](const unsigned short* Xb, const unsigned short* Wt,
                         const float* asrc, const float* adst, const float* bflat,
                         const unsigned short* linWt, const float* linb, float* Yout) {
        // 1) H[h] = X @ W[h]   (WMMA bf16, fp32 accum)
        gemm_wmma_bf16_k<<<dim3((N + 127) / 128, 4), 256, 128 * 128 * 2, stream>>>(
            Xb, Wt, Hbuf, nullptr, N, 128, 128,
            /*aHead*/0L, /*btHead*/(long)128 * 128, /*cHead*/(long)N * 128, 0);
        // 2) attention score dots
        row_dot_k<<<dim3((N + 7) / 8, 4), 256, 0, stream>>>(Hbuf, asrc, adst, ssrc, sdst, N);
        // 3) init accumulators
        zero2_k<<<(4 * N + 255) / 256, 256, 0, stream>>>(denom, mkey, 4 * N);
        init_cat_k<<<(int)(((long)N * 512 + 255) / 256), 256, 0, stream>>>(cat, bflat, (long)N * 512);
        // 4) segment softmax + scatter
        int totalEH = Etot * 4;
        edge_max_k<<<(totalEH + 255) / 256, 256, 0, stream>>>(srcA, dstA, ssrc, sdst, mkey, E, N, totalEH);
        edge_exp_k<<<(totalEH + 255) / 256, 256, 0, stream>>>(srcA, dstA, ssrc, sdst, mkey, denom, ew, E, N, totalEH);
        edge_scatter_k<<<(totalEH + 7) / 8, 256, 0, stream>>>(srcA, dstA, ew, denom, Hbuf, cat, E, N, totalEH);
        // 5) concat -> bf16 -> fused linear + ReLU (WMMA, K=512)
        cvt_bf16_k<<<(int)(((long)N * 512 + 255) / 256), 256, 0, stream>>>(cat, catbf, (long)N * 512);
        gemm_wmma_bf16_k<<<dim3((N + 127) / 128, 1), 256, 512 * 128 * 2, stream>>>(
            catbf, linWt, Yout, linb, N, 512, 128, 0L, 0L, 0L, 1);
    };

    // encoder
    run_layer(Xbf, WtE, enc_asrc, enc_adst, enc_b, LWtE, enc_lin_b, Yenc);
    cvt_bf16_k<<<(N * D + 255) / 256, 256, 0, stream>>>(Yenc, Ybf, (long)N * D);
    // decoder (writes final result into d_out)
    run_layer(Ybf, WtD, dec_asrc, dec_adst, dec_b, LWtD, dec_lin_b, out);
}
